// RelativeMultiheadAttentionMechanism_39324720562898
// MI455X (gfx1250) — compile-verified, matrix operand order checked by reading, not run on hardware
//
#include <hip/hip_runtime.h>

typedef _Float16 half_t;
typedef __attribute__((ext_vector_type(16))) _Float16 v16h;
typedef __attribute__((ext_vector_type(8)))  float    v8f;

#define D_MODEL 512
#define HEADS   8
#define DK      64
#define QLEN    512
#define KLEN    1024

__device__ __forceinline__ v8f wmma16(v16h a, v16h b, v8f c) {
  // v_wmma_f32_16x16x32_f16  D = A*B + C  (f32 accumulate)
  return __builtin_amdgcn_wmma_f32_16x16x32_f16(
      /*neg_a=*/false, a, /*neg_b=*/false, b,
      /*c_mod=*/(short)0, c, /*reuse_a=*/false, /*reuse_b=*/false);
}

// 16-bit A/B fragment packing (ISA 7.12.2): lane = (hlf, lr); element e of the
// v16h holds K = kbase + e + 8*hlf + (e>=8 ? 8 : 0)  -> two contiguous 8-elem chunks.
__device__ __forceinline__ v16h frag_from_f32(const float* __restrict__ rowp,
                                              int hlf, int kbase) {
  const float4* c0 = reinterpret_cast<const float4*>(rowp + kbase + 8 * hlf);
  const float4* c1 = reinterpret_cast<const float4*>(rowp + kbase + 16 + 8 * hlf);
  float4 x0 = c0[0], x1 = c0[1], y0 = c1[0], y1 = c1[1];
  v16h r;
  r[0]=(half_t)x0.x;  r[1]=(half_t)x0.y;  r[2]=(half_t)x0.z;  r[3]=(half_t)x0.w;
  r[4]=(half_t)x1.x;  r[5]=(half_t)x1.y;  r[6]=(half_t)x1.z;  r[7]=(half_t)x1.w;
  r[8]=(half_t)y0.x;  r[9]=(half_t)y0.y;  r[10]=(half_t)y0.z; r[11]=(half_t)y0.w;
  r[12]=(half_t)y1.x; r[13]=(half_t)y1.y; r[14]=(half_t)y1.z; r[15]=(half_t)y1.w;
  return r;
}

__device__ __forceinline__ v16h frag_from_f16(const half_t* __restrict__ rowp,
                                              int hlf, int kbase) {
  union { v16h v; uint4 q[2]; } u;
  u.q[0] = *reinterpret_cast<const uint4*>(rowp + kbase + 8 * hlf);
  u.q[1] = *reinterpret_cast<const uint4*>(rowp + kbase + 16 + 8 * hlf);
  return u.v;
}

// ---------------------------------------------------------------------------
// Generic projection GEMM:  C[m,n] = sum_k A[m,k] * W[n,k]   (X @ W^T)
// 128 threads = 4 waves; each wave owns one 16-col subtile, block = 16x64 tile.
// mode 0: f16 out, [b,h,row,dk] layout          (K / V projections)
// mode 2: f16 out x2 with +bias0/+bias1         (Q -> q+u, q+v)
// mode 3: f16 out, [h,row,dk] layout, batch=1   (P projection)
// mode 4: f32 out, plain [row, n] layout        (output projection)
// ---------------------------------------------------------------------------
__global__ __launch_bounds__(128) void proj_gemm(
    const float* __restrict__ A, long abstride, int arowoff,
    const float* __restrict__ W, int mode,
    void* __restrict__ o0, void* __restrict__ o1,
    const float* __restrict__ bias0, const float* __restrict__ bias1, int R) {
  const int b    = blockIdx.z;
  const int m0   = blockIdx.y * 16;
  const int tid  = threadIdx.x;
  const int wv   = tid >> 5;
  const int lane = tid & 31;
  const int hlf  = lane >> 4;
  const int lr   = lane & 15;
  const int n0   = blockIdx.x * 64 + wv * 16;

  const float* arow = A + (long)b * abstride + (long)(arowoff + m0 + lr) * D_MODEL;
  const float* wrow = W + (long)(n0 + lr) * D_MODEL;

  v8f acc = {};
  #pragma unroll 4
  for (int kk = 0; kk < D_MODEL; kk += 32) {
    v16h af = frag_from_f32(arow, hlf, kk);
    v16h bf = frag_from_f32(wrow, hlf, kk);   // rows of W  ==  B = W^T
    acc = wmma16(af, bf, acc);
  }

  const int n  = n0 + lr;          // C element g: row = m0+g+8*hlf, col = n
  const int h  = n >> 6;
  const int dk = n & 63;
  #pragma unroll
  for (int g = 0; g < 8; ++g) {
    const int row = m0 + g + 8 * hlf;
    const float val = acc[g];
    if (mode == 0) {
      ((half_t*)o0)[(((long)b * HEADS + h) * R + row) * DK + dk] = (half_t)val;
    } else if (mode == 2) {
      const long idx = (((long)b * HEADS + h) * R + row) * DK + dk;
      ((half_t*)o0)[idx] = (half_t)(val + bias0[n]);
      ((half_t*)o1)[idx] = (half_t)(val + bias1[n]);
    } else if (mode == 3) {
      ((half_t*)o0)[((long)h * R + row) * DK + dk] = (half_t)val;
    } else {
      ((float*)o0)[((long)b * R + row) * D_MODEL + n] = val;
    }
  }
}

// ---------------------------------------------------------------------------
// Fused relative attention for one (b, h, 16-query-row strip).
// 128 threads = 4 waves.  LDS: raw-BD strip 16x1024 f32 = 64 KB.
//   pass 1: BD_raw = (q+v)*p^T strip -> LDS
//   pass 2: AC = (q+u)*k^T per 16-col tile; e = (AC + BD_raw[i,|512+i-j|])/8,
//           masked, written straight into the aw output region
//   pass 3: row softmax in-place on aw (global, L2-hot)
//   pass 4: cv = aw @ v via WMMA -> f32 workspace
// ---------------------------------------------------------------------------
__global__ __launch_bounds__(128) void relattn_fused(
    const half_t* __restrict__ kh, const half_t* __restrict__ vh,
    const half_t* __restrict__ quh, const half_t* __restrict__ qvh,
    const half_t* __restrict__ ph, const int* __restrict__ mask,
    float* __restrict__ aw, float* __restrict__ cvw) {
  __shared__ float sBD[16][KLEN];   // exactly 64 KB

  const int b    = blockIdx.z;
  const int h    = blockIdx.y;
  const int i0   = blockIdx.x * 16;
  const int tid  = threadIdx.x;
  const int wv   = tid >> 5;
  const int lane = tid & 31;
  const int hlf  = lane >> 4;
  const int lr   = lane & 15;
  const long bh  = (long)b * HEADS + h;

  // A fragments for this row strip (K = 64 -> two k-steps), loaded once.
  const half_t* qurow = quh + (bh * QLEN + i0 + lr) * DK;
  const half_t* qvrow = qvh + (bh * QLEN + i0 + lr) * DK;
  const v16h qa0 = frag_from_f16(qurow, hlf, 0);
  const v16h qa1 = frag_from_f16(qurow, hlf, 32);
  const v16h va0 = frag_from_f16(qvrow, hlf, 0);
  const v16h va1 = frag_from_f16(qvrow, hlf, 32);

  // ---- pass 1: raw BD strip into LDS ----
  for (int it = 0; it < 16; ++it) {
    const int jt = wv + 4 * it;
    const half_t* prow = ph + ((long)h * KLEN + jt * 16 + lr) * DK;
    v16h p0 = frag_from_f16(prow, hlf, 0);
    v16h p1 = frag_from_f16(prow, hlf, 32);
    v8f acc = {};
    acc = wmma16(va0, p0, acc);
    acc = wmma16(va1, p1, acc);
    #pragma unroll
    for (int g = 0; g < 8; ++g)
      sBD[g + 8 * hlf][jt * 16 + lr] = acc[g];
  }
  __syncthreads();

  // ---- pass 2: AC tile + rel-shift gather + mask + scale -> aw region ----
  float* awrow0 = aw + (bh * QLEN + i0) * KLEN;
  for (int it = 0; it < 16; ++it) {
    const int jt = wv + 4 * it;
    const half_t* krow = kh + (bh * KLEN + jt * 16 + lr) * DK;
    v16h k0 = frag_from_f16(krow, hlf, 0);
    v16h k1 = frag_from_f16(krow, hlf, 32);
    v8f acc = {};
    acc = wmma16(qa0, k0, acc);
    acc = wmma16(qa1, k1, acc);
    const int j = jt * 16 + lr;
    #pragma unroll
    for (int g = 0; g < 8; ++g) {
      const int il = g + 8 * hlf;
      int d = (KLEN - QLEN) + (i0 + il) - j;     // rel_shift index = |512+i-j|
      if (d < 0) d = -d;
      float e = (acc[g] + sBD[il][d]) * 0.125f;  // 1/sqrt(64)
      const int mv = mask[((long)b * QLEN + i0 + il) * KLEN + j];
      if (mv == 0) e = -3.402823466e38f;
      awrow0[(long)il * KLEN + j] = e;
    }
  }
  __threadfence_block();
  __syncthreads();

  // ---- pass 3: row softmax (8 lanes per row, contiguous in one wave) ----
  {
    const int row = tid >> 3, sub = tid & 7;
    float* er = awrow0 + (long)row * KLEN + sub * 128;
    const float4* e4 = (const float4*)er;
    float mx = -3.402823466e38f;
    for (int t = 0; t < 32; ++t) {
      float4 v = e4[t];
      mx = fmaxf(mx, fmaxf(fmaxf(v.x, v.y), fmaxf(v.z, v.w)));
    }
    for (int m = 1; m < 8; m <<= 1) mx = fmaxf(mx, __shfl_xor(mx, m, 8));
    float s = 0.0f;
    for (int t = 0; t < 32; ++t) {
      float4 v = e4[t];
      s += __expf(v.x - mx) + __expf(v.y - mx) + __expf(v.z - mx) + __expf(v.w - mx);
    }
    for (int m = 1; m < 8; m <<= 1) s += __shfl_xor(s, m, 8);
    const float inv = 1.0f / s;
    float4* w4 = (float4*)er;
    for (int t = 0; t < 32; ++t) {
      float4 v = e4[t];
      w4[t] = make_float4(__expf(v.x - mx) * inv, __expf(v.y - mx) * inv,
                          __expf(v.z - mx) * inv, __expf(v.w - mx) * inv);
    }
  }
  __threadfence_block();
  __syncthreads();

  // ---- pass 4: cv = aw @ v ; wave wv owns head cols [wv*16, wv*16+16) ----
  {
    v8f acc = {};
    const half_t* vbase = vh + bh * KLEN * DK;
    const float* awrow = awrow0 + (long)lr * KLEN;    // A row = lr
    for (int kt = 0; kt < 32; ++kt) {
      v16h afr = frag_from_f32(awrow, hlf, kt * 32);
      union { v16h v; unsigned short u[16]; } rb;     // B = v (NN): gather cols
      #pragma unroll
      for (int e = 0; e < 16; ++e) {
        const int K = e + 8 * hlf + (e >= 8 ? 8 : 0);
        rb.u[e] = ((const unsigned short*)vbase)[(kt * 32 + K) * DK + wv * 16 + lr];
      }
      acc = wmma16(afr, rb.v, acc);
    }
    #pragma unroll
    for (int g = 0; g < 8; ++g) {
      const int il = g + 8 * hlf;
      cvw[((long)b * QLEN + i0 + il) * D_MODEL + h * DK + wv * 16 + lr] = acc[g];
    }
  }
}

extern "C" void kernel_launch(void* const* d_in, const int* in_sizes, int n_in,
                              void* d_out, int out_size, void* d_ws, size_t ws_size,
                              hipStream_t stream) {
  const float* key  = (const float*)d_in[0];
  // d_in[1] = query: unused — reference computes q from key[:, -QLEN:]
  const float* pos  = (const float*)d_in[2];
  const int*   mask = (const int*)d_in[3];
  const float* ub   = (const float*)d_in[4];
  const float* vb   = (const float*)d_in[5];
  const float* Wk   = (const float*)d_in[6];
  const float* Wv   = (const float*)d_in[7];
  const float* Wq   = (const float*)d_in[8];
  const float* Wp   = (const float*)d_in[9];
  const float* Wo   = (const float*)d_in[10];

  float* out = (float*)d_out;
  float* aw  = out + (size_t)8 * QLEN * D_MODEL;   // cv first, then aw

  // workspace carve-up (all 16B aligned)
  half_t* khw = (half_t*)d_ws;
  half_t* vhw = khw + (size_t)8 * HEADS * KLEN * DK;      // 4M halves
  half_t* quw = vhw + (size_t)8 * HEADS * KLEN * DK;
  half_t* qvw = quw + (size_t)8 * HEADS * QLEN * DK;      // 2M halves
  half_t* phw = qvw + (size_t)8 * HEADS * QLEN * DK;
  float*  cvw = (float*)(phw + (size_t)HEADS * KLEN * DK);

  const dim3 blk(128);
  // K, V projections: M = 8*1024 rows
  proj_gemm<<<dim3(8, KLEN / 16, 8), blk, 0, stream>>>(
      key, (long)KLEN * D_MODEL, 0, Wk, 0, khw, nullptr, nullptr, nullptr, KLEN);
  proj_gemm<<<dim3(8, KLEN / 16, 8), blk, 0, stream>>>(
      key, (long)KLEN * D_MODEL, 0, Wv, 0, vhw, nullptr, nullptr, nullptr, KLEN);
  // Q projection from key[:, MLEN:], fused +u_bias / +v_bias
  proj_gemm<<<dim3(8, QLEN / 16, 8), blk, 0, stream>>>(
      key, (long)KLEN * D_MODEL, KLEN - QLEN, Wq, 2, quw, qvw, ub, vb, QLEN);
  // P projection (no batch)
  proj_gemm<<<dim3(8, KLEN / 16, 1), blk, 0, stream>>>(
      pos, 0, 0, Wp, 3, phw, nullptr, nullptr, nullptr, KLEN);

  // fused relative attention: 32 row strips x 8 heads x 8 batches
  relattn_fused<<<dim3(QLEN / 16, HEADS, 8), blk, 0, stream>>>(
      khw, vhw, quw, qvw, phw, mask, aw, cvw);

  // output projection: 4096 x 512 @ W_out^T -> d_out
  proj_gemm<<<dim3(8, (8 * QLEN) / 16, 1), blk, 0, stream>>>(
      cvw, 0, 0, Wo, 4, out, nullptr, nullptr, nullptr, 8 * QLEN);

  (void)in_sizes; (void)n_in; (void)out_size; (void)ws_size;
}
